// GAT_75428215652430
// MI455X (gfx1250) — compile-verified, matrix operand order checked by reading, not run on hardware
//
#include <hip/hip_runtime.h>
#include <hip/hip_bf16.h>

typedef __attribute__((ext_vector_type(16))) _Float16 v16h;
typedef __attribute__((ext_vector_type(8)))  _Float16 h8;
typedef __attribute__((ext_vector_type(8)))  float    v8f;
typedef __attribute__((ext_vector_type(4)))  float    f4;

#define NEG_SLOPE 0.2f
#define EPS 1e-16f

// ---- ordered-uint encoding for float atomicMax ----
__device__ __forceinline__ unsigned enc_f(float f) {
    unsigned u = __float_as_uint(f);
    return (u >> 31) ? ~u : (u | 0x80000000u);
}
__device__ __forceinline__ float dec_f(unsigned u) {
    return (u >> 31) ? __uint_as_float(u & 0x7FFFFFFFu) : __uint_as_float(~u);
}

// ---- convert W [K x Ncols] row-major f32 -> Wt [Ncols x K] f16 (col-major) ----
__global__ void gat_convert_wt(const float* __restrict__ W, _Float16* __restrict__ Wt,
                               int K, int Ncols) {
    int i = blockIdx.x * blockDim.x + threadIdx.x;
    if (i >= K * Ncols) return;
    int k = i % K;
    int n = i / K;
    Wt[n * K + k] = (_Float16)W[k * Ncols + n];
}

// ---- WMMA GEMM: C[M x NTILES*16] = A[M x 128] * B[128 x NTILES*16] ----
// One wave owns a 16-row strip: A fragments for all 4 k-steps are loaded once
// and reused across every column tile (B is tiny and L2/WGP$-resident).
template <int NTILES>
__global__ void gat_gemm_wmma(const float* __restrict__ A, const _Float16* __restrict__ Bt,
                              float* __restrict__ C, int M) {
    const int ncols = NTILES * 16;
    int tile_m = blockIdx.x;
    int lane = threadIdx.x;          // 0..31
    int r  = lane & 15;
    int hi = lane >> 4;              // 0 or 1

    int row0 = tile_m * 16;
    bool full_tile = (row0 + 16 <= M);   // wave-uniform
    int arow = row0 + r;
    if (arow >= M) arow = M - 1;         // clamp loads, keep EXEC all-ones

    // Load all four A k-step fragments once.
    // halves 0..7 -> K = k0 + hi*8 + (0..7); halves 8..15 -> +16
    v16h a[4];
#pragma unroll
    for (int ks = 0; ks < 4; ++ks) {
        const float* ap = A + (long long)arow * 128 + ks * 32 + hi * 8;
        f4 a0 = *(const f4*)(ap);
        f4 a1 = *(const f4*)(ap + 4);
        f4 a2 = *(const f4*)(ap + 16);
        f4 a3 = *(const f4*)(ap + 20);
#pragma unroll
        for (int i = 0; i < 4; ++i) {
            a[ks][i]      = (_Float16)a0[i];
            a[ks][4 + i]  = (_Float16)a1[i];
            a[ks][8 + i]  = (_Float16)a2[i];
            a[ks][12 + i] = (_Float16)a3[i];
        }
    }

#pragma unroll
    for (int tn = 0; tn < NTILES; ++tn) {
        int bcol = tn * 16 + r;
        v8f c = {};
#pragma unroll
        for (int ks = 0; ks < 4; ++ks) {
            // B fragment: same K pattern, N in lanes; Bt col-major -> contiguous b128 loads
            const _Float16* bp = Bt + (long long)bcol * 128 + ks * 32 + hi * 8;
            h8 b0 = *(const h8*)(bp);
            h8 b1 = *(const h8*)(bp + 16);
            v16h b;
#pragma unroll
            for (int i = 0; i < 8; ++i) { b[i] = b0[i]; b[8 + i] = b1[i]; }
            c = __builtin_amdgcn_wmma_f32_16x16x32_f16(false, a[ks], false, b,
                                                       (short)0, c, false, false);
        }
        // D: lane holds column tn*16 + (lane&15); VGPR v -> row v + hi*8
        int n = tn * 16 + r;
        if (full_tile) {
            // fast path: unguarded, clause-friendly stores from one base pointer
            float* cp = C + (long long)(row0 + hi * 8) * ncols + n;
#pragma unroll
            for (int v = 0; v < 8; ++v) cp[v * ncols] = c[v];
        } else {
#pragma unroll
            for (int v = 0; v < 8; ++v) {
                int m = row0 + v + hi * 8;
                if (m < M) C[(long long)m * ncols + n] = c[v];
            }
        }
    }
}

// ---- per-node attention dots: as[n,h] = <h1[n,h,:], a_src[h,:]>, same for ad ----
__global__ void gat_alphas(const float* __restrict__ h, const float* __restrict__ a_src,
                           const float* __restrict__ a_dst,
                           float* __restrict__ as, float* __restrict__ ad,
                           int N, int H) {
    int i = blockIdx.x * blockDim.x + threadIdx.x;
    if (i >= N * H) return;
    int n = i / H, hd = i % H;
    const float* hp = h + (long long)n * H * 32 + hd * 32;
    const float* sp = a_src + hd * 32;
    const float* dp = a_dst + hd * 32;
    float s = 0.f, d = 0.f;
#pragma unroll
    for (int c = 0; c < 32; ++c) {
        float v = hp[c];
        s += v * sp[c];
        d += v * dp[c];
    }
    as[i] = s;
    ad[i] = d;
}

// ---- edge pass 1: segment max of leaky_relu(as[src]+ad[dst]) over dst ----
__global__ void gat_edge_max(const int* __restrict__ ei, int E, int N,
                             const float* __restrict__ as, const float* __restrict__ ad,
                             unsigned* __restrict__ mx, int H) {
    int e = blockIdx.x * blockDim.x + threadIdx.x;
    int ET = E + N;
    if (e >= ET) return;
    int s = (e < E) ? ei[e] : (e - E);
    int d = (e < E) ? ei[E + e] : (e - E);
    for (int hd = 0; hd < H; ++hd) {
        float v = as[s * H + hd] + ad[d * H + hd];
        v = v > 0.f ? v : NEG_SLOPE * v;
        atomicMax(&mx[d * H + hd], enc_f(v));
    }
}

// ---- edge pass 2: segment sum of exp(e - max[dst]) ----
__global__ void gat_edge_sum(const int* __restrict__ ei, int E, int N,
                             const float* __restrict__ as, const float* __restrict__ ad,
                             const unsigned* __restrict__ mx, float* __restrict__ sm,
                             int H) {
    int e = blockIdx.x * blockDim.x + threadIdx.x;
    int ET = E + N;
    if (e >= ET) return;
    int s = (e < E) ? ei[e] : (e - E);
    int d = (e < E) ? ei[E + e] : (e - E);
    for (int hd = 0; hd < H; ++hd) {
        float v = as[s * H + hd] + ad[d * H + hd];
        v = v > 0.f ? v : NEG_SLOPE * v;
        atomicAdd(&sm[d * H + hd], expf(v - dec_f(mx[d * H + hd])));
    }
}

// ---- edge pass 3: acc[dst,h,:] += h[src,h,:] * alpha  (one thread per edge*head) ----
__global__ void gat_edge_scatter(const int* __restrict__ ei, int E, int N,
                                 const float* __restrict__ as, const float* __restrict__ ad,
                                 const unsigned* __restrict__ mx, const float* __restrict__ sm,
                                 const float* __restrict__ h, float* __restrict__ acc,
                                 int H) {
    int i = blockIdx.x * blockDim.x + threadIdx.x;
    int tot = (E + N) * H;
    if (i >= tot) return;
    int hd = i % H;
    int e  = i / H;
    int s = (e < E) ? ei[e] : (e - E);
    int d = (e < E) ? ei[E + e] : (e - E);
    float v = as[s * H + hd] + ad[d * H + hd];
    v = v > 0.f ? v : NEG_SLOPE * v;
    float alpha = expf(v - dec_f(mx[d * H + hd])) / (sm[d * H + hd] + EPS);
    const float* hp = h   + (long long)s * H * 32 + hd * 32;
    float*       op = acc + (long long)d * H * 32 + hd * 32;
#pragma unroll
    for (int c = 0; c < 32; c += 4) {
        f4 v4 = *(const f4*)(hp + c);
        atomicAdd(op + c + 0, v4[0] * alpha);
        atomicAdd(op + c + 1, v4[1] * alpha);
        atomicAdd(op + c + 2, v4[2] * alpha);
        atomicAdd(op + c + 3, v4[3] * alpha);
    }
}

// ---- bias + ELU (in place) ----
__global__ void gat_bias_elu(float* __restrict__ a, const float* __restrict__ b,
                             int total, int stride) {
    int i = blockIdx.x * blockDim.x + threadIdx.x;
    if (i >= total) return;
    float v = a[i] + b[i % stride];
    a[i] = v > 0.f ? v : (expf(v) - 1.0f);
}

// ---- bias add (in place) ----
__global__ void gat_bias_add(float* __restrict__ a, const float* __restrict__ b,
                             int total, int stride) {
    int i = blockIdx.x * blockDim.x + threadIdx.x;
    if (i >= total) return;
    a[i] += b[i % stride];
}

extern "C" void kernel_launch(void* const* d_in, const int* in_sizes, int n_in,
                              void* d_out, int out_size, void* d_ws, size_t ws_size,
                              hipStream_t stream) {
    const float* x      = (const float*)d_in[0];
    const int*   ei     = (const int*)d_in[1];
    const float* W1     = (const float*)d_in[2];
    const float* a_src1 = (const float*)d_in[3];
    const float* a_dst1 = (const float*)d_in[4];
    const float* b1     = (const float*)d_in[5];
    const float* W2     = (const float*)d_in[6];
    const float* a_src2 = (const float*)d_in[7];
    const float* a_dst2 = (const float*)d_in[8];
    const float* b2     = (const float*)d_in[9];

    const int FIN = 128, H1 = 4, F1 = H1 * 32; // 128
    const int C2 = 32;
    int N = in_sizes[0] / FIN;
    int E = in_sizes[1] / 2;
    int ET = E + N;

    // carve workspace
    char* p = (char*)d_ws;
    auto carve = [&](size_t bytes) -> void* {
        void* r = (void*)p;
        p += (bytes + 255) & ~(size_t)255;
        return r;
    };
    _Float16* W1t = (_Float16*)carve((size_t)FIN * F1 * sizeof(_Float16));
    _Float16* W2t = (_Float16*)carve((size_t)F1 * C2 * sizeof(_Float16));
    float*    h1   = (float*)carve((size_t)N * F1 * sizeof(float));
    float*    acc1 = (float*)carve((size_t)N * F1 * sizeof(float));
    float*    as1  = (float*)carve((size_t)N * H1 * sizeof(float));
    float*    ad1  = (float*)carve((size_t)N * H1 * sizeof(float));
    unsigned* mx1  = (unsigned*)carve((size_t)N * H1 * sizeof(unsigned));
    float*    sm1  = (float*)carve((size_t)N * H1 * sizeof(float));
    float*    h2   = (float*)carve((size_t)N * C2 * sizeof(float));
    float*    as2  = (float*)carve((size_t)N * sizeof(float));
    float*    ad2  = (float*)carve((size_t)N * sizeof(float));
    unsigned* mx2  = (unsigned*)carve((size_t)N * sizeof(unsigned));
    float*    sm2  = (float*)carve((size_t)N * sizeof(float));
    float*    out  = (float*)d_out;                       // N x 32, also layer-2 accumulator

    const int TB = 256;

    // weight conversion (transposed, f16)
    gat_convert_wt<<<(FIN * F1 + TB - 1) / TB, TB, 0, stream>>>(W1, W1t, FIN, F1);
    gat_convert_wt<<<(F1 * C2 + TB - 1) / TB, TB, 0, stream>>>(W2, W2t, F1, C2);

    // zero/initialize accumulators and softmax stats (enc(x) >= 0 for all finite x)
    hipMemsetAsync(acc1, 0, (size_t)N * F1 * sizeof(float), stream);
    hipMemsetAsync(mx1,  0, (size_t)N * H1 * sizeof(unsigned), stream);
    hipMemsetAsync(sm1,  0, (size_t)N * H1 * sizeof(float), stream);
    hipMemsetAsync(mx2,  0, (size_t)N * sizeof(unsigned), stream);
    hipMemsetAsync(sm2,  0, (size_t)N * sizeof(float), stream);
    hipMemsetAsync(out,  0, (size_t)N * C2 * sizeof(float), stream);

    // ---- layer 1 ----
    gat_gemm_wmma<8><<<(N + 15) / 16, 32, 0, stream>>>(x, W1t, h1, N);
    gat_alphas<<<(N * H1 + TB - 1) / TB, TB, 0, stream>>>(h1, a_src1, a_dst1, as1, ad1, N, H1);
    gat_edge_max<<<(ET + TB - 1) / TB, TB, 0, stream>>>(ei, E, N, as1, ad1, mx1, H1);
    gat_edge_sum<<<(ET + TB - 1) / TB, TB, 0, stream>>>(ei, E, N, as1, ad1, mx1, sm1, H1);
    gat_edge_scatter<<<(ET * H1 + TB - 1) / TB, TB, 0, stream>>>(ei, E, N, as1, ad1, mx1, sm1,
                                                                 h1, acc1, H1);
    gat_bias_elu<<<(N * F1 + TB - 1) / TB, TB, 0, stream>>>(acc1, b1, N * F1, F1);

    // ---- layer 2 ----
    gat_gemm_wmma<2><<<(N + 15) / 16, 32, 0, stream>>>(acc1, W2t, h2, N);
    gat_alphas<<<(N + TB - 1) / TB, TB, 0, stream>>>(h2, a_src2, a_dst2, as2, ad2, N, 1);
    gat_edge_max<<<(ET + TB - 1) / TB, TB, 0, stream>>>(ei, E, N, as2, ad2, mx2, 1);
    gat_edge_sum<<<(ET + TB - 1) / TB, TB, 0, stream>>>(ei, E, N, as2, ad2, mx2, sm2, 1);
    gat_edge_scatter<<<(ET + TB - 1) / TB, TB, 0, stream>>>(ei, E, N, as2, ad2, mx2, sm2,
                                                            h2, out, 1);
    gat_bias_add<<<(N * C2 + TB - 1) / TB, TB, 0, stream>>>(out, b2, N * C2, C2);
}
